// FP8Linear_56006373540395
// MI455X (gfx1250) — matrix operand order repaired and checked by reading, not compiled, not run on hardware
//
#include <hip/hip_runtime.h>

typedef __bf16 bhalf;
typedef __attribute__((ext_vector_type(16))) bhalf v16bf;
typedef __attribute__((ext_vector_type(8)))  float v8f;
typedef __attribute__((ext_vector_type(4)))  unsigned int v4u;
typedef __attribute__((ext_vector_type(4)))  int v4i;

using as1_v4i = __attribute__((address_space(1))) v4i;
using as3_v4i = __attribute__((address_space(3))) v4i;

static constexpr int KDIM = 2048;   // IN
static constexpr int NDIM = 2048;   // OUT
static constexpr int BM = 128, BN = 128, BK = 64;
static constexpr int LDA  = BK + 8;      // 72 bf16 = 144B row stride -> conflict-free b128 LDS reads
static constexpr int NBUF = 4;           // power-of-two ring -> constant buffer offsets under unroll-4
static constexpr int NIT  = KDIM / BK;   // 32 (divisible by 4)
static constexpr int TILE_ELEMS = BM * LDA;

union FragAB { v16bf v; v4u u[2]; unsigned short s[16]; };

#if __has_builtin(__builtin_amdgcn_global_load_async_to_lds_b128)
#define HAVE_ASYNC_LDS 1
#else
#define HAVE_ASYNC_LDS 0
#endif

__device__ __forceinline__ void stage_b128(const bhalf* g, bhalf* l) {
#if HAVE_ASYNC_LDS
    __builtin_amdgcn_global_load_async_to_lds_b128(
        (as1_v4i*)g, (as3_v4i*)l, 0, 0);
#else
    *(v4u*)l = *(const v4u*)g;   // fallback: reg staging
#endif
}

__device__ __forceinline__ void wait_stage_keep4() {
#if HAVE_ASYNC_LDS
    asm volatile("s_wait_asynccnt 0x4" ::: "memory");   // newest tile (4 ops) may remain in flight
#endif
}
__device__ __forceinline__ void wait_stage_all() {
#if HAVE_ASYNC_LDS
    asm volatile("s_wait_asynccnt 0x0" ::: "memory");
#endif
}

// fp8 e4m3fn -> f32 (exact)
__device__ __forceinline__ float fp8_e4m3_to_f32(unsigned b) {
    unsigned sgn = b >> 7;
    unsigned e = (b >> 3) & 0xF;
    unsigned m = b & 0x7;
    float v;
    if (e == 0) {
        v = (float)m * 0x1p-9f;                       // subnormal
    } else if (e == 15 && m == 7) {
        v = __builtin_nanf("");                       // e4m3fn NaN
    } else {
        unsigned bits = ((e + 120u) << 23) | (m << 20);
        v = __builtin_bit_cast(float, bits);
    }
    return sgn ? -v : v;
}

// ---------------- Kernel 1: dequantize weights fp8 -> bf16*scale ----------------
__global__ __launch_bounds__(256) void fp8lin_dequant_w(
    const unsigned char* __restrict__ w8, const float* __restrict__ scale,
    bhalf* __restrict__ wbf, int n)
{
    int i = (blockIdx.x * blockDim.x + threadIdx.x) * 8;
    if (i >= n) return;
    float scl = (float)(bhalf)(*scale);               // reference casts scale to bf16 first
    uint2 pack = *(const uint2*)(w8 + i);
    union { v4u u; unsigned short s[8]; } o;
    #pragma unroll
    for (int e = 0; e < 8; ++e) {
        unsigned word = (e < 4) ? pack.x : pack.y;
        unsigned b = (word >> (8 * (e & 3))) & 0xFF;
        bhalf h = (bhalf)(fp8_e4m3_to_f32(b) * scl);  // product exact in f32; single RNE round
        o.s[e] = __builtin_bit_cast(unsigned short, h);
    }
    *(v4u*)(wbf + i) = o.u;
}

// ---------------- B-fragment loader (lane = column N, 16 contiguous K elems) ----------------
template <bool INLINE_DEQ>
__device__ __forceinline__ void load_bfrags(
    FragAB bf[2][2], const void* __restrict__ W, const size_t bro[2],
    int k0, int half, float scl)
{
    #pragma unroll
    for (int j = 0; j < 2; ++j) {
        #pragma unroll
        for (int s = 0; s < 2; ++s) {
            size_t base = bro[j] + k0 + s * 32 + half * 16;
            if (INLINE_DEQ) {
                v4u raw = *(const v4u*)((const unsigned char*)W + base);  // 16 fp8 bytes
                #pragma unroll
                for (int w = 0; w < 4; ++w) {
                    unsigned word = raw[w];
                    #pragma unroll
                    for (int e = 0; e < 4; ++e) {
                        unsigned b = (word >> (8 * e)) & 0xFF;
                        bhalf h = (bhalf)(fp8_e4m3_to_f32(b) * scl);
                        bf[j][s].s[w * 4 + e] = __builtin_bit_cast(unsigned short, h);
                    }
                }
            } else {
                const bhalf* wb = (const bhalf*)W + base;
                bf[j][s].u[0] = *(const v4u*)(wb);
                bf[j][s].u[1] = *(const v4u*)(wb + 8);
            }
        }
    }
}

// ---------------- one K=64 compute step (two K=32 WMMA sub-steps) ----------------
__device__ __forceinline__ void compute_step(
    v8f acc[4][2], const bhalf* As, int cb, const int afr[4], int half,
    const FragAB b[2][2])
{
    #pragma unroll
    for (int s = 0; s < 2; ++s) {
        FragAB af[4];
        #pragma unroll
        for (int i = 0; i < 4; ++i) {
            const bhalf* ap = &As[cb + afr[i] + s * 32 + half * 16];
            af[i].u[0] = *(const v4u*)(ap);
            af[i].u[1] = *(const v4u*)(ap + 8);
        }
        #pragma unroll
        for (int i = 0; i < 4; ++i)
            #pragma unroll
            for (int j = 0; j < 2; ++j)
                acc[i][j] = __builtin_amdgcn_wmma_f32_16x16x32_bf16(
                    false, af[i].v, false, b[j][s].v, (short)0, acc[i][j], false, false);
    }
}

// ---------------- Kernel 2: bf16 WMMA GEMM  out = x @ wdq^T + bias ----------------
template <bool INLINE_DEQ>
__global__ __launch_bounds__(256) void fp8lin_gemm(
    const bhalf* __restrict__ X,      // (M, K) bf16
    const void*  __restrict__ W,      // (N, K) bf16 (pre-dequant) or fp8 bytes
    const float* __restrict__ scale,
    const bhalf* __restrict__ bias,   // (N)
    bhalf* __restrict__ out)          // (M, N)
{
    __shared__ bhalf As[NBUF * TILE_ELEMS];

    const int tid  = threadIdx.x;
    const int lane = tid & 31;
    const int wave = tid >> 5;
    const int wm   = wave >> 2;       // 0..1  -> 64 rows each
    const int wn   = wave & 3;        // 0..3  -> 32 cols each
    const int half = lane >> 4;       // K-half select: this code's contiguous k = 16*half + s
    const int l16  = lane & 15;

    const int m0 = blockIdx.y * BM;
    const int n0 = blockIdx.x * BN;

    // A staging: 128 rows x 64 cols = 1024 x b128 chunks; thread t -> rows (t>>3)+{0,32,64,96}, col (t&7)*8
    const int row_st = tid >> 3;          // 0..31
    const int cc_st  = (tid & 7) * 8;
    const bhalf* gA = X + (size_t)(m0 + row_st) * KDIM + cc_st;
    bhalf*       lA = &As[row_st * LDA + cc_st];

    // fragment row bases in LDS for the 4 M-tiles of this wave
    int afr[4];
    #pragma unroll
    for (int i = 0; i < 4; ++i) afr[i] = (wm * 64 + i * 16 + l16) * LDA;

    // weight row (= output column) element bases for the 2 N-tiles
    size_t bro[2];
    #pragma unroll
    for (int j = 0; j < 2; ++j) bro[j] = (size_t)(n0 + wn * 32 + j * 16 + l16) * KDIM;

    float scl = 0.f;
    if (INLINE_DEQ) scl = (float)(bhalf)(*scale);

    v8f acc[4][2];
    #pragma unroll
    for (int i = 0; i < 4; ++i)
        #pragma unroll
        for (int j = 0; j < 2; ++j)
            #pragma unroll
            for (int e = 0; e < 8; ++e) acc[i][j][e] = 0.f;

    // ---- prologue: stage tiles 0 and 1 into buffers 0 and 1 ----
    #pragma unroll
    for (int q = 0; q < 4; ++q)
        stage_b128(gA + (size_t)q * 32 * KDIM, lA + q * 32 * LDA);
    #pragma unroll
    for (int q = 0; q < 4; ++q)
        stage_b128(gA + (size_t)q * 32 * KDIM + BK, lA + TILE_ELEMS + q * 32 * LDA);
    wait_stage_keep4();                 // tile 0 complete (in-order retirement)
    __syncthreads();

    FragAB b0[2][2], b1[2][2];
    load_bfrags<INLINE_DEQ>(b0, W, bro, 0, half, scl);

    // one pipeline body: consume buffer `cb` with fragments `bc`; stage tile itc+2
    // into `stg` (constant offset); prefetch B fragments for itc+1 into `bp`.
    auto body = [&](int itc, int cb, bhalf* stg, FragAB (&bc)[2][2], FragAB (&bp)[2][2]) {
        const bool more = (itc + 2) < NIT;
        if (more) {
            const int nk = (itc + 2) * BK;
            #pragma unroll
            for (int q = 0; q < 4; ++q)
                stage_b128(gA + (size_t)q * 32 * KDIM + nk, stg + q * 32 * LDA);
        }
        if (itc + 1 < NIT)
            load_bfrags<INLINE_DEQ>(bp, W, bro, (itc + 1) * BK, half, scl);

        compute_step(acc, As, cb, afr, half, bc);

        if (more) wait_stage_keep4();
        else      wait_stage_all();
        __syncthreads();
    };

    for (int it = 0; it < NIT; it += 4) {
        body(it + 0, 0 * TILE_ELEMS, lA + 2 * TILE_ELEMS, b0, b1);
        body(it + 1, 1 * TILE_ELEMS, lA + 3 * TILE_ELEMS, b1, b0);
        body(it + 2, 2 * TILE_ELEMS, lA + 0 * TILE_ELEMS, b0, b1);
        body(it + 3, 3 * TILE_ELEMS, lA + 1 * TILE_ELEMS, b1, b0);
    }

    // ---- epilogue: + bias, round to bf16, store ----
    #pragma unroll
    for (int j = 0; j < 2; ++j) {
        const int n = n0 + wn * 32 + j * 16 + l16;
        const float bv = (float)bias[n];
        #pragma unroll
        for (int i = 0; i < 4; ++i) {
            const int mbase = m0 + wm * 64 + i * 16 + half * 8;
            #pragma unroll
            for (int r = 0; r < 8; ++r) {
                out[(size_t)(mbase + r) * NDIM + n] = (bhalf)(acc[i][j][r] + bv);
            }
        }
    }
}

extern "C" void kernel_launch(void* const* d_in, const int* in_sizes, int n_in,
                              void* d_out, int out_size, void* d_ws, size_t ws_size,
                              hipStream_t stream) {
    const bhalf*         x     = (const bhalf*)d_in[0];
    const unsigned char* w8    = (const unsigned char*)d_in[1];
    const float*         scale = (const float*)d_in[2];
    const bhalf*         bias  = (const bhalf*)d_in[3];
    bhalf*               out   = (bhalf*)d_out;

    const int wcount = in_sizes[1];                       // 2048*2048 fp8 elems
    const int M = (int)((size_t)in_sizes[0] / KDIM);      // 32768 rows
    dim3 grid(NDIM / BN, M / BM);

    if (ws_size >= (size_t)wcount * sizeof(bhalf)) {
        bhalf* wdq = (bhalf*)d_ws;
        int blocks = (wcount + 256 * 8 - 1) / (256 * 8);
        fp8lin_dequant_w<<<blocks, 256, 0, stream>>>(w8, scale, wdq, wcount);
        fp8lin_gemm<false><<<grid, 256, 0, stream>>>(x, wdq, scale, bias, out);
    } else {
        fp8lin_gemm<true><<<grid, 256, 0, stream>>>(x, w8, scale, bias, out);
    }
}
